// Guided_Attn_15015205667031
// MI455X (gfx1250) — compile-verified, matrix operand order checked by reading, not run on hardware
//
#include <hip/hip_runtime.h>
#include <hip/hip_bf16.h>

#if __has_builtin(__builtin_amdgcn_global_load_async_to_lds_b128) && \
    __has_builtin(__builtin_amdgcn_global_load_async_to_lds_b64)
#define USE_ASYNC_LDS 1
#else
#define USE_ASYNC_LDS 0
#endif

typedef __attribute__((ext_vector_type(2)))  float  v2f;
typedef __attribute__((ext_vector_type(8)))  float  v8f;
typedef __attribute__((ext_vector_type(2)))  __bf16 v2bf;
typedef __attribute__((ext_vector_type(8)))  __bf16 v8bf;
typedef __attribute__((ext_vector_type(16))) __bf16 v16bf;

// generic vector-of-int types matching the async-LDS builtin signatures
typedef int avi2 __attribute__((vector_size(8)));
typedef int avi4 __attribute__((vector_size(16)));
typedef __attribute__((address_space(1))) avi2 g2i;   // global int2
typedef __attribute__((address_space(3))) avi2 l2i;   // LDS int2
typedef __attribute__((address_space(1))) avi4 g4i;   // global int4
typedef __attribute__((address_space(3))) avi4 l4i;   // LDS int4

#define B_       8
#define C_       64
#define DQK_     8
#define N_       4096   // H*W
#define TM       64     // key/value tile width (2x K=32 bf16 WMMA)
#define RPB      128    // query rows per block (8 waves x 16 rows)
#define NTHREADS 256
#define NTILES   (N_ / TM)

__device__ __forceinline__ v16bf cat8(v8bf lo, v8bf hi) {
  v16bf r;
#pragma unroll
  for (int i = 0; i < 8; ++i) { r[i] = lo[i]; r[i + 8] = hi[i]; }
  return r;
}

// ---- DPP16 cross-lane reductions over each 16-lane group (VALU only) ----
// fused VOP2+DPP16: one instruction per butterfly step
__device__ __forceinline__ float red_max16(float x) {
  float t;
  asm("v_max_num_f32_dpp %0, %1, %1 quad_perm:[1,0,3,2] row_mask:0xf bank_mask:0xf bound_ctrl:1\n\t"
      "v_max_num_f32_dpp %0, %0, %0 quad_perm:[2,3,0,1] row_mask:0xf bank_mask:0xf bound_ctrl:1\n\t"
      "v_max_num_f32_dpp %0, %0, %0 row_half_mirror row_mask:0xf bank_mask:0xf bound_ctrl:1\n\t"
      "v_max_num_f32_dpp %0, %0, %0 row_mirror row_mask:0xf bank_mask:0xf bound_ctrl:1"
      : "=&v"(t) : "v"(x));
  return t;
}
__device__ __forceinline__ float red_add16(float x) {
  float t;
  asm("v_add_f32_dpp %0, %1, %1 quad_perm:[1,0,3,2] row_mask:0xf bank_mask:0xf bound_ctrl:1\n\t"
      "v_add_f32_dpp %0, %0, %0 quad_perm:[2,3,0,1] row_mask:0xf bank_mask:0xf bound_ctrl:1\n\t"
      "v_add_f32_dpp %0, %0, %0 row_half_mirror row_mask:0xf bank_mask:0xf bound_ctrl:1\n\t"
      "v_add_f32_dpp %0, %0, %0 row_mirror row_mask:0xf bank_mask:0xf bound_ctrl:1"
      : "=&v"(t) : "v"(x));
  return t;
}

__device__ __forceinline__ void wait_async_lds() {
#if USE_ASYNC_LDS
#if __has_builtin(__builtin_amdgcn_s_wait_asynccnt)
  __builtin_amdgcn_s_wait_asynccnt(0);
#else
  asm volatile("s_wait_asynccnt 0x0" ::: "memory");
#endif
#endif
}

// ---------------- 1x1 conv projections ----------------
// k is written PAIR-INTERLEAVED: k_ws[b][o/2][n][2] so the energy-WMMA B
// operand {k[d][col], k[d+1][col]} is one contiguous 8-byte LDS load.
__global__ __launch_bounds__(NTHREADS)
void proj_qk_kernel(const float* __restrict__ query,
                    const float* __restrict__ Wq, const float* __restrict__ bq,
                    const float* __restrict__ Wk, const float* __restrict__ bk,
                    float* __restrict__ q_ws, float* __restrict__ k_ws) {
  int idx = blockIdx.x * blockDim.x + threadIdx.x;
  if (idx >= B_ * DQK_ * N_) return;
  int n = idx & (N_ - 1);
  int o = (idx >> 12) & (DQK_ - 1);
  int b = idx >> 15;
  const float* qcol = query + (size_t)b * C_ * N_ + n;
  float aq = bq[o], ak = bk[o];
#pragma unroll 8
  for (int c = 0; c < C_; ++c) {
    float x = qcol[(size_t)c * N_];
    aq = fmaf(Wq[o * C_ + c], x, aq);
    ak = fmaf(Wk[o * C_ + c], x, ak);
  }
  q_ws[idx] = aq;
  k_ws[(((size_t)b * (DQK_ / 2) + (o >> 1)) * N_ + n) * 2 + (o & 1)] = ak;
}

// v is written with a column permutation inside each 64-wide tile:
// n = t*64 + g*16 + j  ->  n' = t*64 + (g>>1)*32 + 2j + (g&1)
// (P columns use the same permutation, softmax is order-invariant over m)
__global__ __launch_bounds__(NTHREADS)
void proj_v_kernel(const float* __restrict__ value,
                   const float* __restrict__ Wv, const float* __restrict__ bv,
                   __bf16* __restrict__ v_ws) {
  int idx = blockIdx.x * blockDim.x + threadIdx.x;
  if (idx >= B_ * C_ * N_) return;
  int n  = idx & (N_ - 1);
  int co = (idx >> 12) & (C_ - 1);
  int b  = idx >> 18;
  const float* vcol = value + (size_t)b * C_ * N_ + n;
  float a = bv[co];
#pragma unroll 8
  for (int c = 0; c < C_; ++c)
    a = fmaf(Wv[co * C_ + c], vcol[(size_t)c * N_], a);
  int t = n >> 6, g = (n >> 4) & 3, j = n & 15;
  int np = (t << 6) | ((g & 2) << 4) | (j << 1) | (g & 1);
  v_ws[((size_t)b * C_ + co) * N_ + np] = (__bf16)a;
}

// ---------------- fused flash attention ----------------
__global__ __launch_bounds__(NTHREADS)
void attn_kernel(const float* __restrict__ q_ws, const float* __restrict__ k_ws,
                 const __bf16* __restrict__ v_ws, const float* __restrict__ value,
                 float* __restrict__ out) {
  __shared__ float q_buf[RPB][DQK_];                               //  4 KB
  __shared__ __align__(16) float  kp_lds[2][DQK_ / 2][TM][2];      //  4 KB (double buffered)
  __shared__ __align__(16) __bf16 v_lds[2][C_][TM];                // 16 KB (double buffered)
  __shared__ __align__(16) __bf16 p_buf[NTHREADS / 32][16][TM];    // 16 KB

  const int tid  = threadIdx.x;
  const int wave = tid >> 5;
  const int lane = tid & 31;
  const int hf   = lane >> 4;   // which 16-lane half
  const int lrow = lane & 15;
  const int b    = blockIdx.y;
  const int n0   = blockIdx.x * RPB;
  const int wr0  = wave * 16;

  // per-thread slices of the cooperative tile staging
  const int kj = tid >> 6, kc = tid & 63;          // k: 4 pair-rows x 64 cols (8B / thread)
  const int vc = tid >> 2, vm = (tid & 3) * 16;    // v: 64x64 bf16 (16 bf16 / thread)
  const float*  kg = k_ws + (((size_t)b * (DQK_ / 2) + kj) * N_ + kc) * 2;
  const __bf16* vg = v_ws + ((size_t)b * C_ + vc) * N_ + vm;

  auto stage = [&](int m0, int buf) {
#if USE_ASYNC_LDS
    __builtin_amdgcn_global_load_async_to_lds_b64(
        (g2i*)(kg + (size_t)m0 * 2), (l2i*)(&kp_lds[buf][kj][kc][0]), 0, 0);
    __builtin_amdgcn_global_load_async_to_lds_b128(
        (g4i*)(vg + m0), (l4i*)(&v_lds[buf][vc][vm]), 0, 0);
    __builtin_amdgcn_global_load_async_to_lds_b128(
        (g4i*)(vg + m0 + 8), (l4i*)(&v_lds[buf][vc][vm + 8]), 0, 0);
#else
    v2f  kr = *reinterpret_cast<const v2f*>(kg + (size_t)m0 * 2);
    v8bf v0 = *reinterpret_cast<const v8bf*>(vg + m0);
    v8bf v1 = *reinterpret_cast<const v8bf*>(vg + m0 + 8);
    *reinterpret_cast<v2f*>(&kp_lds[buf][kj][kc][0])   = kr;
    *reinterpret_cast<v8bf*>(&v_lds[buf][vc][vm])      = v0;
    *reinterpret_cast<v8bf*>(&v_lds[buf][vc][vm + 8])  = v1;
#endif
  };

  // stage tile 0 + this block's 128x8 q rows
  stage(0, 0);
  for (int i = tid; i < RPB * DQK_; i += NTHREADS) {
    int d = i >> 7, row = i & (RPB - 1);
    q_buf[row][d] = q_ws[((size_t)b * DQK_ + d) * N_ + n0 + row];
  }
  wait_async_lds();
  __syncthreads();

  // A operand (f32 16x4 layout): lane holds row M=lrow; VGPR0={K0|K2}, VGPR1={K1|K3}
  const int rowg = wr0 + lrow;
  const int d0 = hf ? 2 : 0;
  v2f qa0, qa1;
  qa0.x = q_buf[rowg][d0];     qa0.y = q_buf[rowg][d0 + 1];
  qa1.x = q_buf[rowg][d0 + 4]; qa1.y = q_buf[rowg][d0 + 5];

  float rmax[8], rsum[8];   // rsum is a per-lane PARTIAL sum (reduced in epilogue)
#pragma unroll
  for (int r = 0; r < 8; ++r) { rmax[r] = -3.0e38f; rsum[r] = 0.f; }
  v8f acc[4] = {};            // 16 rows x 64 channels, f32

  const int kb0 = hf ? 8 : 0; // bf16 A/B layout: K-offset per lane half

  for (int it = 0; it < NTILES; ++it) {
    const int cur = it & 1;
    if (it + 1 < NTILES) stage((it + 1) * TM, cur ^ 1);  // overlap with compute

    // ---- energy: four 16x16 f32 subtiles via chained V_WMMA_F32_16X16X4_F32 ----
    // B operand = one aligned ds_load_b64 per WMMA (pair-interleaved k)
    v8f e[4];
#pragma unroll
    for (int s = 0; s < 4; ++s) {
      int col = s * 16 + lrow;
      v8f t = {};
      v2f kb0v = *reinterpret_cast<const v2f*>(&kp_lds[cur][hf][col][0]);
      t = __builtin_amdgcn_wmma_f32_16x16x4_f32(false, qa0, false, kb0v, (short)0, t, false, false);
      v2f kb1v = *reinterpret_cast<const v2f*>(&kp_lds[cur][2 + hf][col][0]);
      t = __builtin_amdgcn_wmma_f32_16x16x4_f32(false, qa1, false, kb1v, (short)0, t, false, false);
      e[s] = t;
    }

    // ---- online softmax; rows live in 16-lane halves of the C layout ----
    float sc8[8];
#pragma unroll
    for (int r = 0; r < 8; ++r) {
      float mx = red_max16(fmaxf(fmaxf(e[0][r], e[1][r]), fmaxf(e[2][r], e[3][r])));
      float mnew = fmaxf(rmax[r], mx);
      float corr = __expf(rmax[r] - mnew);
      rmax[r] = mnew;
      sc8[r] = corr;
      float p0 = __expf(e[0][r] - mnew);
      float p1 = __expf(e[1][r] - mnew);
      float p2 = __expf(e[2][r] - mnew);
      float p3 = __expf(e[3][r] - mnew);
      rsum[r] = rsum[r] * corr + ((p0 + p1) + (p2 + p3));  // lane-local partial sum
      // stage P in the sigma-permuted column order: two packed b32 stores
      v2bf pk01; pk01[0] = (__bf16)p0; pk01[1] = (__bf16)p1;
      v2bf pk23; pk23[0] = (__bf16)p2; pk23[1] = (__bf16)p3;
      *reinterpret_cast<v2bf*>(&p_buf[wave][hf * 8 + r][2 * lrow])      = pk01;
      *reinterpret_cast<v2bf*>(&p_buf[wave][hf * 8 + r][32 + 2 * lrow]) = pk23;
    }

    // rescale output accumulators
#pragma unroll
    for (int ch = 0; ch < 4; ++ch)
#pragma unroll
      for (int r = 0; r < 8; ++r)
        acc[ch][r] *= sc8[r];

    // ---- P as two bf16 16x32 A operands (K windows 0..31 and 32..63) ----
    v16bf pa[2];
#pragma unroll
    for (int h = 0; h < 2; ++h) {
      int base = h * 32 + kb0;
      v8bf plo = *reinterpret_cast<const v8bf*>(&p_buf[wave][lrow][base]);
      v8bf phi = *reinterpret_cast<const v8bf*>(&p_buf[wave][lrow][base + 16]);
      pa[h] = cat8(plo, phi);
    }

    // ---- PV: 4 chunks of 16 channels x 2 K-halves ----
#pragma unroll
    for (int ch = 0; ch < 4; ++ch) {
      int c = ch * 16 + lrow;   // B operand: lane holds column c, K striped like A
#pragma unroll
      for (int h = 0; h < 2; ++h) {
        int base = h * 32 + kb0;
        v8bf vlo = *reinterpret_cast<const v8bf*>(&v_lds[cur][c][base]);
        v8bf vhi = *reinterpret_cast<const v8bf*>(&v_lds[cur][c][base + 16]);
        v16bf vb = cat8(vlo, vhi);
        acc[ch] = __builtin_amdgcn_wmma_f32_16x16x32_bf16(false, pa[h], false, vb,
                                                          (short)0, acc[ch], false, false);
      }
    }

    wait_async_lds();   // next tile fully landed in LDS
    __syncthreads();    // single barrier per iteration (double buffered)
  }

  // ---- epilogue: reduce partial row sums once, normalize, residual, store ----
  float inv[8];
#pragma unroll
  for (int r = 0; r < 8; ++r) inv[r] = 1.0f / red_add16(rsum[r]);
#pragma unroll
  for (int ch = 0; ch < 4; ++ch) {
#pragma unroll
    for (int r = 0; r < 8; ++r) {
      int c = ch * 16 + lrow;
      int n = n0 + wr0 + hf * 8 + r;
      size_t off = ((size_t)b * C_ + c) * N_ + n;
      out[off] = acc[ch][r] * inv[r] + value[off];
    }
  }
}

extern "C" void kernel_launch(void* const* d_in, const int* in_sizes, int n_in,
                              void* d_out, int out_size, void* d_ws, size_t ws_size,
                              hipStream_t stream) {
  const float* query = (const float*)d_in[0];
  const float* value = (const float*)d_in[1];
  const float* Wq    = (const float*)d_in[2];
  const float* bq    = (const float*)d_in[3];
  const float* Wk    = (const float*)d_in[4];
  const float* bk    = (const float*)d_in[5];
  const float* Wv    = (const float*)d_in[6];
  const float* bv    = (const float*)d_in[7];
  float* out = (float*)d_out;

  float*  q_ws = (float*)d_ws;                             // 1 MB
  float*  k_ws = q_ws + (size_t)B_ * DQK_ * N_;            // 1 MB
  __bf16* v_ws = (__bf16*)(k_ws + (size_t)B_ * DQK_ * N_); // 4 MB

  proj_qk_kernel<<<(B_ * DQK_ * N_ + NTHREADS - 1) / NTHREADS, NTHREADS, 0, stream>>>(
      query, Wq, bq, Wk, bk, q_ws, k_ws);
  proj_v_kernel<<<(B_ * C_ * N_ + NTHREADS - 1) / NTHREADS, NTHREADS, 0, stream>>>(
      value, Wv, bv, v_ws);

  dim3 grid(N_ / RPB, B_);
  attn_kernel<<<grid, NTHREADS, 0, stream>>>(q_ws, k_ws, v_ws, value, out);
}
